// Attention_67972152426705
// MI455X (gfx1250) — compile-verified, matrix operand order checked by reading, not run on hardware
//
#include <hip/hip_runtime.h>

typedef __attribute__((ext_vector_type(16))) _Float16 v16h;
typedef __attribute__((ext_vector_type(8)))  _Float16 v8h;
typedef __attribute__((ext_vector_type(8)))  float    v8f;

#define B_ 32
#define S_ 1024
#define D_ 512
// 1/sqrt(512)
#define SCALE_INV 0.044194173824159216f

// LDS carve-up (dynamic shared memory). qh aliases ph (disjoint lifetimes).
#define OFF_SCORES 0                        // 16*1024 f32 = 65536 B
#define OFF_PH     (16 * S_ * 4)            // 16*1024 f16 = 32768 B
#define OFF_QH     OFF_PH                   // 16*512  f16 = 16384 B (alias)
#define OFF_RED    (OFF_PH + 16 * S_ * 2)   // 128 f32
#define OFF_RMAX   (OFF_RED + 128 * 4)      // 16 f32
#define OFF_RINV   (OFF_RMAX + 16 * 4)      // 16 f32
#define OFF_RMF    (OFF_RINV + 16 * 4)      // 1024 f32 = 4096 B
#define SMEM_BYTES (OFF_RMF + S_ * 4)       // ~103 KB -> 3 blocks / 320KB WGP

__global__ __launch_bounds__(128)
void attn_masked_sm_kernel(const float* __restrict__ Q,
                           const float* __restrict__ K,
                           const float* __restrict__ V,
                           const int*   __restrict__ RM,
                           float* __restrict__ Out,
                           float* __restrict__ Attn)
{
    extern __shared__ char smem[];
    float*    scores = (float*)(smem + OFF_SCORES);     // [16][1024] masked_vec
    _Float16* ph     = (_Float16*)(smem + OFF_PH);      // [16][1024] P in f16
    _Float16* qh     = (_Float16*)(smem + OFF_QH);      // [16][512]  Q in f16
    float*    red    = (float*)(smem + OFF_RED);        // [128]
    float*    rowmax = (float*)(smem + OFF_RMAX);       // [16]
    float*    rowinv = (float*)(smem + OFF_RINV);       // [16]
    float*    rmf    = (float*)(smem + OFF_RMF);        // [1024] mask as float

    const int tid  = threadIdx.x;
    const int lane = tid & 31;
    const int wave = tid >> 5;
    const int b    = blockIdx.x >> 6;            // 64 q-tiles per batch
    const int q0   = (blockIdx.x & 63) << 4;

    const float* Qb = Q + ((size_t)b * S_ + q0) * D_;
    const float* Kb = K + (size_t)b * S_ * D_;
    const float* Vb = V + (size_t)b * S_ * D_;

    // ---- stage rep_mask (as float) into LDS, then Q tile (f32 -> f16) ----
    for (int i = tid; i < S_; i += 128)
        rmf[i] = (RM[(size_t)b * S_ + i] != 0) ? 1.0f : 0.0f;
    for (int i = tid; i < 16 * D_; i += 128)
        qh[i] = (_Float16)Qb[i];
    __syncthreads();

    const int half  = (lane < 16) ? 0 : 8;   // K-offset split per half-wave
    const int ln    = lane & 15;
    const int rbase = (lane < 16) ? 0 : 8;   // C/D row base per half-wave

    // hoist per-row mask values (branch-free, loaded once)
    float rmqv[8];
#pragma unroll
    for (int r = 0; r < 8; ++r)
        rmqv[r] = rmf[q0 + rbase + r];

    // ======== phase 1: masked_vec = (Q K^T / sqrt(D)) * mask ========
    for (int kt = wave * 16; kt < wave * 16 + 16; ++kt) {
        const int k0 = kt << 4;
        v8f c = {};
        const float* kp = Kb + (size_t)(k0 + ln) * D_;  // B lane = key position
        for (int ks = 0; ks < D_ / 32; ++ks) {
            const int kb = ks * 32;
            // A fragment (16-bit A layout: elems 0-7 -> K=kb+half+0..7, 8-15 -> +16)
            const _Float16* ap = qh + ln * D_ + kb + half;
            v8h alo = *(const v8h*)ap;
            v8h ahi = *(const v8h*)(ap + 16);
            v16h a;
#pragma unroll
            for (int e = 0; e < 8; ++e) { a[e] = alo[e]; a[e + 8] = ahi[e]; }
            // B fragment from global K: contiguous in d for this lane's key row
            const float* bp = kp + kb + half;
            float4 f0 = *(const float4*)(bp);
            float4 f1 = *(const float4*)(bp + 4);
            float4 f2 = *(const float4*)(bp + 16);
            float4 f3 = *(const float4*)(bp + 20);
            float tf[16] = { f0.x, f0.y, f0.z, f0.w, f1.x, f1.y, f1.z, f1.w,
                             f2.x, f2.y, f2.z, f2.w, f3.x, f3.y, f3.z, f3.w };
            v16h bfr;
#pragma unroll
            for (int e = 0; e < 16; ++e) bfr[e] = (_Float16)tf[e];

            c = __builtin_amdgcn_wmma_f32_16x16x32_f16(
                    false, a, false, bfr, (short)0, c, false, false);
        }
        // scale + multiplicative mask (branch-free), store masked_vec to LDS
        const int   kglob = k0 + ln;
        const float rmk   = rmf[kglob];
#pragma unroll
        for (int r = 0; r < 8; ++r) {
            const int   row  = rbase + r;
            const float tri  = (kglob < q0 + row) ? 1.0f : 0.0f;
            const float mbit = rmk * rmqv[r] * tri;
            scores[row * S_ + kglob] = c[r] * SCALE_INV * mbit;
        }
    }
    __syncthreads();

    // ======== phase 2: faithful masked softmax statistics ========
    {
        const int row = tid >> 3;     // 16 rows x 8 threads
        const int sub = tid & 7;
        float pm = -3.4e38f;
        for (int j = 0; j < S_ / 8; ++j)
            pm = fmaxf(pm, scores[row * S_ + sub + j * 8]);
        red[tid] = pm;
        __syncthreads();
        if (tid < 16) {
            float m = red[tid * 8];
            for (int j = 1; j < 8; ++j) m = fmaxf(m, red[tid * 8 + j]);
            rowmax[tid] = m;
        }
        __syncthreads();
        const float rmax  = rowmax[row];
        const int   qglob = q0 + row;
        const float rmq   = rmf[qglob];
        float ps = 0.0f;
        for (int j = 0; j < S_ / 8; ++j) {
            const int   col  = sub + j * 8;
            const float tri  = (col < qglob) ? 1.0f : 0.0f;
            const float mbit = rmq * rmf[col] * tri;
            ps += __expf(scores[row * S_ + col] - rmax) * mbit;
        }
        red[tid] = ps;
        __syncthreads();
        if (tid < 16) {
            float s = 0.0f;
            for (int j = 0; j < 8; ++j) s += red[tid * 8 + j];
            s = s + ((s == 0.0f) ? 1.0f : 0.0f) + 1e-20f;
            rowinv[tid] = 1.0f / s;
        }
        __syncthreads();
    }

    // ======== phase 3: write attn_sm (coalesced) + keep f16 P in LDS ========
    // NOTE: overwrites the qh alias region -- qh is dead after phase 1.
    for (int idx = tid; idx < 16 * S_; idx += 128) {
        const int   row   = idx >> 10;
        const int   col   = idx & (S_ - 1);
        const int   qglob = q0 + row;
        const float tri   = (col < qglob) ? 1.0f : 0.0f;
        const float mbit  = rmf[qglob] * rmf[col] * tri;
        const float av = __expf(scores[idx] - rowmax[row]) * mbit * rowinv[row];
        Attn[((size_t)b * S_ + qglob) * S_ + col] = av;
        ph[idx] = (_Float16)av;
    }
    __syncthreads();

    // ======== phase 4: out = P @ V ========
    for (int nt = wave * 8; nt < wave * 8 + 8; ++nt) {
        const int n0 = nt << 4;
        v8f c = {};
        for (int ks = 0; ks < S_ / 32; ++ks) {
            const int kb = ks * 32;
            const _Float16* ap = ph + ln * S_ + kb + half;
            v8h alo = *(const v8h*)ap;
            v8h ahi = *(const v8h*)(ap + 16);
            v16h a;
#pragma unroll
            for (int e = 0; e < 8; ++e) { a[e] = alo[e]; a[e + 8] = ahi[e]; }
            // B fragment from V: lane = output column, elems walk key positions
            const float* vp = Vb + (size_t)(kb + half) * D_ + n0 + ln;
            v16h bfr;
#pragma unroll
            for (int e = 0; e < 8; ++e) {
                bfr[e]     = (_Float16)vp[(size_t)e * D_];
                bfr[e + 8] = (_Float16)vp[(size_t)(e + 16) * D_];
            }
            c = __builtin_amdgcn_wmma_f32_16x16x32_f16(
                    false, a, false, bfr, (short)0, c, false, false);
        }
        float* op = Out + ((size_t)b * S_ + q0) * D_ + n0 + ln;
#pragma unroll
        for (int r = 0; r < 8; ++r)
            op[(size_t)(rbase + r) * D_] = c[r];
    }
}

extern "C" void kernel_launch(void* const* d_in, const int* in_sizes, int n_in,
                              void* d_out, int out_size, void* d_ws, size_t ws_size,
                              hipStream_t stream) {
    (void)in_sizes; (void)n_in; (void)d_ws; (void)ws_size; (void)out_size;
    const float* Q  = (const float*)d_in[0];
    const float* K  = (const float*)d_in[1];
    const float* V  = (const float*)d_in[2];
    const int*   RM = (const int*)d_in[3];
    float* Out  = (float*)d_out;                       // [B,S,D]
    float* Attn = Out + (size_t)B_ * S_ * D_;          // [B,S,S]

    // allow >64KB dynamic LDS (CDNA5 WGP has 320KB); idempotent, capture-safe
    (void)hipFuncSetAttribute((const void*)attn_masked_sm_kernel,
                              hipFuncAttributeMaxDynamicSharedMemorySize,
                              SMEM_BYTES);

    dim3 grid(B_ * (S_ / 16));
    dim3 block(128);
    attn_masked_sm_kernel<<<grid, block, SMEM_BYTES, stream>>>(Q, K, V, RM, Out, Attn);
}
